// SelfAttention_37211596652837
// MI455X (gfx1250) — matrix operand order repaired and checked by reading, not compiled, not run on hardware
//
#include <hip/hip_runtime.h>
#include <hip/hip_bf16.h>
#include <math.h>

typedef __bf16 bf16;
typedef __attribute__((ext_vector_type(16))) __bf16 v16bf;
typedef __attribute__((ext_vector_type(8)))  __bf16 v8bf;
typedef __attribute__((ext_vector_type(8)))  float  v8f;
typedef __attribute__((ext_vector_type(4)))  int    v4i;

#define BB 4
#define TT 2048
#define CC 512
#define HH 8
#define HD 64

#if __has_builtin(__builtin_amdgcn_global_load_async_to_lds_b128)
#define HAVE_ASYNC 1
#else
#define HAVE_ASYNC 0
#endif

// async-to-LDS builtin takes (int4 AS1* src, int4 AS3* dst, imm offset, imm cpol)
#define GPTR(p) ((__attribute__((address_space(1))) v4i*)(p))
#define LPTR(p) ((__attribute__((address_space(3))) v4i*)(p))

__device__ __forceinline__ void wait_async0() {
#if __has_builtin(__builtin_amdgcn_s_wait_asynccnt)
  __builtin_amdgcn_s_wait_asynccnt(0);
#else
  asm volatile("s_wait_asynccnt 0" ::: "memory");
#endif
}

// Scheduling hints: issue a group of DS-reads (mask 0x100) before a group of
// WMMAs (mask 0x008) so LDS latency pipelines under partial s_wait_dscnt.
#if __has_builtin(__builtin_amdgcn_sched_group_barrier)
#define SCHED_DS_READS(n) __builtin_amdgcn_sched_group_barrier(0x100, (n), 0)
#define SCHED_WMMAS(n)    __builtin_amdgcn_sched_group_barrier(0x008, (n), 0)
#else
#define SCHED_DS_READS(n)
#define SCHED_WMMAS(n)
#endif

#define WMMA_BF16(a, b, c) \
  __builtin_amdgcn_wmma_f32_16x16x32_bf16(false, (a), false, (b), (short)0, \
                                          (c), false, false)

// A-frag (16x32 bf16) per-lane: elements 0..7 -> K = half*16 + e,
// elements 8..15 -> K = half*16 + e   (two contiguous 16B runs)
// B-frag (32x16 bf16) per-lane: element e -> K = half*16 + e (one 32B run)

// ---------------------------------------------------------------------------
// Kernel 1: qkv = x @ Wqkv + bqkv.
//   Q, K -> bf16 [B,H,T,HD] (Q pre-scaled by 1/sqrt(HD))
//   V    -> bf16 [B,H,HD,T] (transposed, so PV B-frags are contiguous)
// ---------------------------------------------------------------------------
__global__ __launch_bounds__(256)
void qkv_gemm(const float* __restrict__ x, const float* __restrict__ Wqkv,
              const float* __restrict__ bqkv,
              bf16* __restrict__ qws, bf16* __restrict__ kws,
              bf16* __restrict__ vtws) {
  __shared__ __align__(64) bf16 lBt[2][64 * 32];  // [n_local][k], 4KB each

  const int tid  = threadIdx.x;
  const int lane = tid & 31;
  const int wave = tid >> 5;
  const int half = lane >> 4;
  const int l16  = lane & 15;
  const int m0 = (blockIdx.y * 8 + wave) * 16;  // row tile in [B*T]
  const int n0 = blockIdx.x * 64;               // 64-wide col tile in [3C]

  const int nl   = tid & 63;  // staging column 0..63
  const int kseg = tid >> 6;  // staging k-segment 0..3

  // Stage chunk 0 (transposed, bf16-packed)
  {
    const float* wp = Wqkv + (size_t)(kseg * 8) * (3 * CC) + n0 + nl;
    union { bf16 h[8]; uint4 u; } pk;
#pragma unroll
    for (int i = 0; i < 8; ++i) pk.h[i] = (bf16)wp[(size_t)i * (3 * CC)];
    *(uint4*)&lBt[0][nl * 32 + kseg * 8] = pk.u;
  }

  v8f acc[4] = {};
  for (int j = 0; j < CC / 32; ++j) {
    const int buf = j & 1;
    __syncthreads();  // publish stage(j); all waves done reading lBt[1-buf]
    if (j + 1 < CC / 32) {
      const float* wp =
          Wqkv + (size_t)((j + 1) * 32 + kseg * 8) * (3 * CC) + n0 + nl;
      union { bf16 h[8]; uint4 u; } pk;
#pragma unroll
      for (int i = 0; i < 8; ++i) pk.h[i] = (bf16)wp[(size_t)i * (3 * CC)];
      *(uint4*)&lBt[1 - buf][nl * 32 + kseg * 8] = pk.u;
    }
    // A fragment: two contiguous 8-float runs, cvt to bf16
    const float* xr = x + (size_t)(m0 + l16) * CC + j * 32;
    v16bf a;
#pragma unroll
    for (int e = 0; e < 8; ++e) {
      a[e]     = (bf16)xr[half * 16 + e];
      a[8 + e] = (bf16)xr[half * 16 + 8 + e];
    }
    // Load all 4 B-frags (8 ds_load_b128), then 4 back-to-back WMMAs
    v16bf bf4[4];
#pragma unroll
    for (int t = 0; t < 4; ++t)
      bf4[t] = *(const v16bf*)&lBt[buf][(t * 16 + l16) * 32 + half * 16];
#pragma unroll
    for (int t = 0; t < 4; ++t) acc[t] = WMMA_BF16(a, bf4[t], acc[t]);
    SCHED_DS_READS(8);
    SCHED_WMMAS(4);
  }

  // Epilogue: n0 is 64-aligned, so head / q-k-v selector are block constants.
  const int which = n0 / CC;           // 0=Q, 1=K, 2=V
  const int hblk  = (n0 % CC) / HD;    // head index
  const int b_  = m0 >> 11;            // batch (constant over the 16-row tile)
  const int tq0 = (m0 & (TT - 1)) + half * 8;
  if (which == 2) {
    // V: transposed store [B,H,HD,T]; r = 0..7 are consecutive T positions ->
    // pack into one 16B store per tile.
#pragma unroll
    for (int t = 0; t < 4; ++t) {
      const int d = t * 16 + l16;
      const float bias = bqkv[n0 + d];
      union { bf16 h[8]; uint4 u; } pk;
#pragma unroll
      for (int r = 0; r < 8; ++r) pk.h[r] = (bf16)(acc[t][r] + bias);
      *(uint4*)&vtws[((size_t)(b_ * HH + hblk) * HD + d) * TT + tq0] = pk.u;
    }
  } else {
    bf16* dst = (which == 0) ? qws : kws;
    const float scale = (which == 0) ? 0.125f : 1.0f;  // 1/sqrt(HD) into Q
#pragma unroll
    for (int t = 0; t < 4; ++t) {
      const int d = t * 16 + l16;
      const float bias = bqkv[n0 + d];
#pragma unroll
      for (int r = 0; r < 8; ++r) {
        dst[((size_t)(b_ * HH + hblk) * TT + tq0 + r) * HD + d] =
            (bf16)((acc[t][r] + bias) * scale);
      }
    }
  }
}

// ---------------------------------------------------------------------------
// Kernel 2: flash attention per (b,h); 8 waves x 16 queries per workgroup.
// K [T,HD] and V^T [HD,T] chunks staged via async global->LDS, double buffered.
// ALiBi slope recovered from alibias[0,h,0,1] == -m_h.
// ---------------------------------------------------------------------------
__global__ __launch_bounds__(256)
void attn(const bf16* __restrict__ qws, const bf16* __restrict__ kws,
          const bf16* __restrict__ vtws, const float* __restrict__ alibias,
          bf16* __restrict__ attws) {
  __shared__ __align__(64) bf16 lK[2][32 * HD];    // [key][d]   2 x 4 KB
  __shared__ __align__(64) bf16 lVt[2][HD * 32];   // [d][key]   2 x 4 KB
  __shared__ __align__(64) bf16 lP[8][16 * 32];    // per-wave P transpose 8 KB

  const int tid  = threadIdx.x;
  const int lane = tid & 31;
  const int wave = tid >> 5;
  const int half = lane >> 4;
  const int l16  = lane & 15;

  const int qblocks = TT / 128;
  const int qb = blockIdx.x % qblocks;
  const int bh = blockIdx.x / qblocks;
  const int h  = bh % HH;
  const int q0 = qb * 128 + wave * 16;
  const size_t base = (size_t)bh * TT * HD;
  const bf16* gK = kws + base;
  const bf16* gV = vtws + base;  // [HD][T]

  const float nm = alibias[(size_t)h * TT * TT + 1];  // == -m_h

  const int skk  = tid >> 3;        // K staging row (key) 0..31
  const int sko  = (tid & 7) * 8;   // K staging col offset (8 bf16 = 16B)
  const int svd  = tid >> 2;        // V staging row (d) 0..63
  const int svo  = (tid & 3) * 8;   // V staging col offset (8 bf16 = 16B)

  // Stage chunk 0
  {
#if HAVE_ASYNC
    __builtin_amdgcn_global_load_async_to_lds_b128(
        GPTR(gK + (size_t)skk * HD + sko), LPTR(&lK[0][skk * HD + sko]), 0, 0);
    __builtin_amdgcn_global_load_async_to_lds_b128(
        GPTR(gV + (size_t)svd * TT + svo), LPTR(&lVt[0][svd * 32 + svo]), 0, 0);
#else
    *(uint4*)&lK[0][skk * HD + sko] = *(const uint4*)&gK[(size_t)skk * HD + sko];
    *(uint4*)&lVt[0][svd * 32 + svo] = *(const uint4*)&gV[(size_t)svd * TT + svo];
#endif
  }

  // Q fragments (scale folded in): two contiguous 16B runs each
  v16bf qf[2];
#pragma unroll
  for (int kc = 0; kc < 2; ++kc) {
    const bf16* qr = qws + base + (size_t)(q0 + l16) * HD + kc * 32;
    v8bf qlo = *(const v8bf*)&qr[half * 16];
    v8bf qhi = *(const v8bf*)&qr[half * 16 + 8];
#pragma unroll
    for (int e = 0; e < 8; ++e) { qf[kc][e] = qlo[e]; qf[kc][8 + e] = qhi[e]; }
  }

  v8f o[4] = {};
  float rmax[8], rsum[8];
#pragma unroll
  for (int r = 0; r < 8; ++r) { rmax[r] = -1e30f; rsum[r] = 0.0f; }

  for (int j = 0; j < TT / 32; ++j) {
    const int cur = j & 1;
#if HAVE_ASYNC
    wait_async0();  // this wave's copies into buffers[cur] are done
#endif
    __syncthreads();  // all waves' copies visible; all done reading buf[1-cur]
    if (j + 1 < TT / 32) {
      const int ky1 = (j + 1) * 32;
#if HAVE_ASYNC
      __builtin_amdgcn_global_load_async_to_lds_b128(
          GPTR(gK + (size_t)(ky1 + skk) * HD + sko),
          LPTR(&lK[1 - cur][skk * HD + sko]), 0, 0);
      __builtin_amdgcn_global_load_async_to_lds_b128(
          GPTR(gV + (size_t)svd * TT + ky1 + svo),
          LPTR(&lVt[1 - cur][svd * 32 + svo]), 0, 0);
#else
      *(uint4*)&lK[1 - cur][skk * HD + sko] =
          *(const uint4*)&gK[(size_t)(ky1 + skk) * HD + sko];
      *(uint4*)&lVt[1 - cur][svd * 32 + svo] =
          *(const uint4*)&gV[(size_t)svd * TT + ky1 + svo];
#endif
    }
    const int ky = j * 32;

    // S = Q @ K^T : load all 4 K B-frags (8 ds reads), then 4 WMMAs.
    v8f s[2] = {};
    {
      v16bf kf[4];
#pragma unroll
      for (int t = 0; t < 2; ++t)
#pragma unroll
        for (int kc = 0; kc < 2; ++kc)
          kf[t * 2 + kc] = *(const v16bf*)
              &lK[cur][(t * 16 + l16) * HD + kc * 32 + half * 16];
#pragma unroll
      for (int t = 0; t < 2; ++t)
#pragma unroll
        for (int kc = 0; kc < 2; ++kc)
          s[t] = WMMA_BF16(qf[kc], kf[t * 2 + kc], s[t]);
      SCHED_DS_READS(8);
      SCHED_WMMAS(4);
    }

    // Online softmax across the 32 new columns.
#pragma unroll
    for (int r = 0; r < 8; ++r) {
      int m = half * 8 + r;
      float qq  = (float)(q0 + m);
      float k0i = (float)(ky + l16);
      float s0 = s[0][r] + nm * fabsf(qq - k0i);
      float s1 = s[1][r] + nm * fabsf(qq - (k0i + 16.0f));
      float cmax = fmaxf(s0, s1);
#pragma unroll
      for (int msk = 1; msk < 16; msk <<= 1)
        cmax = fmaxf(cmax, __shfl_xor(cmax, msk, 32));
      float mnew  = fmaxf(rmax[r], cmax);
      float alpha = __expf(rmax[r] - mnew);
      rmax[r] = mnew;
      float p0 = __expf(s0 - mnew);
      float p1 = __expf(s1 - mnew);
      float ps = p0 + p1;
#pragma unroll
      for (int msk = 1; msk < 16; msk <<= 1)
        ps += __shfl_xor(ps, msk, 32);
      rsum[r] = rsum[r] * alpha + ps;
#pragma unroll
      for (int t = 0; t < 4; ++t) o[t][r] *= alpha;
      lP[wave][m * 32 + l16]      = (bf16)p0;
      lP[wave][m * 32 + 16 + l16] = (bf16)p1;
    }
    asm volatile("s_wait_dscnt 0" ::: "memory");  // wave-private LDS RAW

    // O += P @ V : P frag (2 reads) + 4 V B-frags (8 reads), then 4 WMMAs.
    v16bf pf;
    {
      v8bf plo = *(const v8bf*)&lP[wave][l16 * 32 + half * 16];
      v8bf phi = *(const v8bf*)&lP[wave][l16 * 32 + half * 16 + 8];
#pragma unroll
      for (int e = 0; e < 8; ++e) { pf[e] = plo[e]; pf[8 + e] = phi[e]; }
    }
    v16bf vf4[4];
#pragma unroll
    for (int t = 0; t < 4; ++t)
      vf4[t] = *(const v16bf*)&lVt[cur][(t * 16 + l16) * 32 + half * 16];
#pragma unroll
    for (int t = 0; t < 4; ++t) o[t] = WMMA_BF16(pf, vf4[t], o[t]);
    SCHED_DS_READS(10);
    SCHED_WMMAS(4);
  }

  // Normalize and write back in [B,T,C] layout (c = h*HD + d) as bf16.
  const int b_ = bh / HH;
#pragma unroll
  for (int r = 0; r < 8; ++r) {
    int m  = half * 8 + r;
    int tq = q0 + m;
    float inv = 1.0f / rsum[r];
#pragma unroll
    for (int nt = 0; nt < 4; ++nt) {
      int c = h * HD + nt * 16 + l16;
      attws[((size_t)b_ * TT + tq) * CC + c] = (bf16)(o[nt][r] * inv);
    }
  }
}

// ---------------------------------------------------------------------------
// Kernel 3: out = att @ Wproj + bproj (fp32 out); same LDS-staged scheme.
// ---------------------------------------------------------------------------
__global__ __launch_bounds__(256)
void proj_gemm(const bf16* __restrict__ attws, const float* __restrict__ Wproj,
               const float* __restrict__ bproj, float* __restrict__ out) {
  __shared__ __align__(64) bf16 lBt[2][64 * 32];

  const int tid  = threadIdx.x;
  const int lane = tid & 31;
  const int wave = tid >> 5;
  const int half = lane >> 4;
  const int l16  = lane & 15;
  const int m0 = (blockIdx.y * 8 + wave) * 16;
  const int n0 = blockIdx.x * 64;

  const int nl   = tid & 63;
  const int kseg = tid >> 6;

  {
    const float* wp = Wproj + (size_t)(kseg * 8) * CC + n0 + nl;
    union { bf16 h[8]; uint4 u; } pk;
#pragma unroll
    for (int i = 0; i < 8; ++i) pk.h[i] = (bf16)wp[(size_t)i * CC];
    *(uint4*)&lBt[0][nl * 32 + kseg * 8] = pk.u;
  }

  v8f acc[4] = {};
  for (int j = 0; j < CC / 32; ++j) {
    const int buf = j & 1;
    __syncthreads();
    if (j + 1 < CC / 32) {
      const float* wp = Wproj + (size_t)((j + 1) * 32 + kseg * 8) * CC + n0 + nl;
      union { bf16 h[8]; uint4 u; } pk;
#pragma unroll
      for (int i = 0; i < 8; ++i) pk.h[i] = (bf16)wp[(size_t)i * CC];
      *(uint4*)&lBt[1 - buf][nl * 32 + kseg * 8] = pk.u;
    }
    const bf16* ar = attws + (size_t)(m0 + l16) * CC + j * 32;
    v16bf a;
    {
      v8bf alo = *(const v8bf*)&ar[half * 16];
      v8bf ahi = *(const v8bf*)&ar[half * 16 + 8];
#pragma unroll
      for (int e = 0; e < 8; ++e) { a[e] = alo[e]; a[8 + e] = ahi[e]; }
    }
    v16bf bf4[4];
#pragma unroll
    for (int t = 0; t < 4; ++t)
      bf4[t] = *(const v16bf*)&lBt[buf][(t * 16 + l16) * 32 + half * 16];
#pragma unroll
    for (int t = 0; t < 4; ++t) acc[t] = WMMA_BF16(a, bf4[t], acc[t]);
    SCHED_DS_READS(8);
    SCHED_WMMAS(4);
  }

#pragma unroll
  for (int t = 0; t < 4; ++t) {
    const int n = n0 + t * 16 + l16;
    const float bias = bproj[n];
#pragma unroll
    for (int r = 0; r < 8; ++r) {
      int m = m0 + half * 8 + r;
      out[(size_t)m * CC + n] = acc[t][r] + bias;
    }
  }
}

extern "C" void kernel_launch(void* const* d_in, const int* in_sizes, int n_in,
                              void* d_out, int out_size, void* d_ws,
                              size_t ws_size, hipStream_t stream) {
  const float* x       = (const float*)d_in[0];
  const float* Wqkv    = (const float*)d_in[1];
  const float* bqkv    = (const float*)d_in[2];
  const float* Wproj   = (const float*)d_in[3];
  const float* bproj   = (const float*)d_in[4];
  const float* alibias = (const float*)d_in[5];

  const size_t qkvElems = (size_t)BB * HH * TT * HD;  // 4M elems (8MB bf16)
  bf16* qws   = (bf16*)d_ws;
  bf16* kws   = qws + qkvElems;
  bf16* vtws  = kws + qkvElems;   // [B,H,HD,T]
  bf16* attws = vtws + qkvElems;  // [B,T,C] bf16; total ws use = 32MB

  dim3 blk(256);
  // QKV: N=1536 -> 24 blocks of 64; M=8192 -> 64 blocks of 128
  qkv_gemm<<<dim3(24, 64), blk, 0, stream>>>(x, Wqkv, bqkv, qws, kws, vtws);
  // Attention: B*H * (T/128) workgroups
  attn<<<dim3(BB * HH * (TT / 128)), blk, 0, stream>>>(qws, kws, vtws, alibias,
                                                       attws);
  // Proj: N=512 -> 8 blocks of 64; M -> 64 blocks
  proj_gemm<<<dim3(8, 64), blk, 0, stream>>>(attws, Wproj, bproj,
                                             (float*)d_out);
}
